// Speller_59064390254771
// MI455X (gfx1250) — matrix operand order repaired and checked by reading, not compiled, not run on hardware
//
#include <hip/hip_runtime.h>
#include <hip/hip_bf16.h>

// Speller (LAS decoder) for MI455X / gfx1250.
// f16 WMMA for all GEMMs; fragment-native packed weights (2x b128 per B-frag);
// f16 KV cache sized to stay resident in the 192MB L2; fused attention+output head.

typedef _Float16 v16h __attribute__((ext_vector_type(16)));
typedef float    v8f  __attribute__((ext_vector_type(8)));
typedef _Float16 h8v  __attribute__((ext_vector_type(8)));
typedef float    f4v  __attribute__((ext_vector_type(4)));

static constexpr int L_  = 2048;
static constexpr int N_  = 32;
static constexpr int T_  = 200;
static constexpr int V_  = 34;
static constexpr int IN_ = 512;
static constexpr int H_  = 256;
static constexpr int Q_  = 256;

__device__ __forceinline__ float sigm(float x) { return 1.0f / (1.0f + __expf(-x)); }

// ---- WMMA fragment loaders (CDNA5 16-bit layouts, wave32) ----
// A: 16x32 (MxK). lanes 0-15: M=lane, K={0..7,16..23}; lanes 16-31: M=lane-16, K={8..15,24..31}
__device__ __forceinline__ v16h load_a16(const _Float16* base, int ld, int lane) {
  int m  = lane & 15;
  int kb = (lane >> 4) << 3;
  const h8v* r = (const h8v*)(base + m * ld + kb);
  h8v lo = r[0];
  h8v hi = r[2];  // +16 halves
  v16h a;
#pragma unroll
  for (int j = 0; j < 8; ++j) { a[j] = lo[j]; a[8 + j] = hi[j]; }
  return a;
}

__device__ __forceinline__ v16h load_a32cvt(const float* base, int ld, int lane) {
  int m  = lane & 15;
  int kb = (lane >> 4) << 3;
  const f4v* r = (const f4v*)(base + (size_t)m * ld + kb);
  f4v x0 = r[0], x1 = r[1], x2 = r[4], x3 = r[5];  // floats 0..7 and 16..23
  v16h a;
#pragma unroll
  for (int j = 0; j < 4; ++j) {
    a[j]      = (_Float16)x0[j];
    a[4 + j]  = (_Float16)x1[j];
    a[8 + j]  = (_Float16)x2[j];
    a[12 + j] = (_Float16)x3[j];
  }
  return a;
}

// B: packed fragment-native layout. Tile (kt,nt) occupies 512 halves; lane's 16 halves contiguous.
__device__ __forceinline__ v16h load_bp(const _Float16* tileBase, int lane) {
  const h8v* r = (const h8v*)(tileBase + lane * 16);
  h8v lo = r[0], hi = r[1];
  v16h b;
#pragma unroll
  for (int j = 0; j < 8; ++j) { b[j] = lo[j]; b[8 + j] = hi[j]; }
  return b;
}

// ------------------- setup kernels -------------------

// Pack virtual B (Ktot x Ncols), B[k][n] = (k<K1 ? srcA[n*K1+k] : srcB[n*K2+(k-K1)]),
// into fragment-native tiles: dst[((kt*NT+nt)*32+lane)*16+j] with
// k = kt*32 + (lane>>4)*8 + (j&7) + (j>>3)*16, n = nt*16 + (lane&15).
__global__ void pack_b_kernel(_Float16* __restrict__ dst,
                              const float* __restrict__ srcA, int K1,
                              const float* __restrict__ srcB, int K2,
                              int Ncols) {
  int Ktot = K1 + K2;
  int total = Ktot * Ncols;
  int idx = blockIdx.x * 256 + threadIdx.x;
  if (idx >= total) return;
  int j    = idx & 15;
  int lane = (idx >> 4) & 31;
  int rest = idx >> 9;
  int NT = Ncols >> 4;
  int nt = rest % NT;
  int kt = rest / NT;
  int k = kt * 32 + ((lane >> 4) << 3) + (j & 7) + ((j >> 3) << 4);
  int n = nt * 16 + (lane & 15);
  float v = (k < K1) ? srcA[n * K1 + k] : srcB[n * K2 + (k - K1)];
  dst[idx] = (_Float16)v;
}

// dst[k*R + j] = (f16) src[j*C + k]   (src is R rows x C cols) — for the scalar output head
__global__ void transpose_f16_kernel(_Float16* __restrict__ dst, const float* __restrict__ src,
                                     int R, int C) {
  int idx = blockIdx.x * 256 + threadIdx.x;
  if (idx >= R * C) return;
  int j = idx / C, k = idx - j * C;
  dst[k * R + j] = (_Float16)src[idx];
}

__global__ void add_bias_kernel(float* __restrict__ dst, const float* __restrict__ a,
                                const float* __restrict__ b, int n) {
  int i = blockIdx.x * 256 + threadIdx.x;
  if (i < n) dst[i] = a[i] + b[i];
}

__global__ void emb_gather_kernel(_Float16* __restrict__ dst, const float* __restrict__ emb,
                                  const int* __restrict__ label_in) {
  int idx = blockIdx.x * 256 + threadIdx.x;
  if (idx >= T_ * N_ * H_) return;
  int t = idx / (N_ * H_);
  int rem = idx - t * N_ * H_;
  int n = rem >> 8, h = rem & 255;
  int lab = label_in[n * T_ + t];
  dst[idx] = (_Float16)emb[lab * H_ + h];
}

__global__ void state_init_kernel(_Float16* __restrict__ hst, float* __restrict__ cst,
                                  const float* __restrict__ inith, const float* __restrict__ initc) {
  int idx = blockIdx.x * 256 + threadIdx.x;
  if (idx >= 3 * N_ * H_) return;
  int i = idx / (N_ * H_);
  int h = idx & 255;
  hst[idx] = (_Float16)inith[i * H_ + h];
  cst[idx] = initc[i * H_ + h];
}

__global__ void q_init_kernel(_Float16* __restrict__ qout, const float* __restrict__ inith2,
                              const float* __restrict__ Wq, const float* __restrict__ bq) {
  int n = blockIdx.x, j = threadIdx.x;
  float acc = bq[j];
  for (int k = 0; k < H_; ++k) acc += inith2[k] * Wq[j * H_ + k];
  qout[n * Q_ + j] = (_Float16)acc;
}

// ------------------- KV projection (WMMA) -------------------
// rows r = l*32+n over 65536 -> 4096 M-tiles; one wave per M-tile, 16 N-tiles x {key,val}
__global__ void kv_proj_kernel(const float* __restrict__ seqs,
                               const _Float16* __restrict__ WkP, const _Float16* __restrict__ WvP,
                               const float* __restrict__ bk, const float* __restrict__ bv,
                               _Float16* __restrict__ key, _Float16* __restrict__ val) {
  const int wave = threadIdx.x >> 5, lane = threadIdx.x & 31;
  const int mtile = blockIdx.x * 8 + wave;           // 0..4095
  const float* Arow0 = seqs + (size_t)mtile * 16 * IN_;
  for (int which = 0; which < 2; ++which) {
    const _Float16* W = which ? WvP : WkP;
    const float* bias = which ? bv : bk;
    _Float16* out = which ? val : key;
    for (int nt = 0; nt < 16; ++nt) {
      v8f acc = {};
      for (int k0 = 0; k0 < IN_; k0 += 32) {
        v16h a = load_a32cvt(Arow0 + k0, IN_, lane);
        v16h b = load_bp(W + ((size_t)((k0 >> 5) * 16 + nt) << 9), lane);
        acc = __builtin_amdgcn_wmma_f32_16x16x32_f16(false, a, false, b, (short)0, acc,
                                                     false, false);
      }
      int col = nt * 16 + (lane & 15);
      float bia = bias[col];
#pragma unroll
      for (int r = 0; r < 8; ++r) {
        int row = mtile * 16 + r + ((lane >> 4) << 3);
        int l = row >> 5, n = row & 31;
        out[((size_t)n * L_ + l) * 256 + col] = (_Float16)(acc[r] + bia);
      }
    }
  }
}

// ------------------- fused LSTM stack + q projection (WMMA, 1 workgroup) -------------------
__global__ void lstm_step_kernel(const _Float16* __restrict__ ctx,
                                 const _Float16* __restrict__ embt,
                                 _Float16* __restrict__ hst, float* __restrict__ cst,
                                 const _Float16* __restrict__ W0, const _Float16* __restrict__ W1,
                                 const _Float16* __restrict__ W2, const _Float16* __restrict__ WqP,
                                 const float* __restrict__ bc0, const float* __restrict__ bc1,
                                 const float* __restrict__ bc2, const float* __restrict__ bq,
                                 _Float16* __restrict__ qout) {
  extern __shared__ char smem[];
  _Float16* Abuf = (_Float16*)smem;                                  // 32 x 768 f16 (48 KB)
  float* gates   = (float*)(smem + 32 * 768 * sizeof(_Float16));     // 32 x 1024 f32 (128 KB)
  const int tid = threadIdx.x;
  const int wave = tid >> 5, lane = tid & 31;

  // A0 = [ctx | x_t | h0]  (32 x 768)
  for (int idx = tid; idx < 32 * 768; idx += 256) {
    int n = idx / 768, k = idx - n * 768;
    _Float16 v;
    if (k < 256)      v = ctx[n * 256 + k];
    else if (k < 512) v = embt[n * 256 + (k - 256)];
    else              v = hst[n * 256 + (k - 512)];
    Abuf[idx] = v;
  }
  __syncthreads();

  for (int cell = 0; cell < 3; ++cell) {
    const _Float16* W = (cell == 0) ? W0 : ((cell == 1) ? W1 : W2);
    const float* bc   = (cell == 0) ? bc0 : ((cell == 1) ? bc1 : bc2);
    const int K = (cell == 0) ? 768 : 512;

    // gates(32x1024) = A(32xK) @ W(Kx1024);  8 waves x (2 Mtiles x 8 Ntiles); NT=64
    for (int tt = 0; tt < 16; ++tt) {
      int mt = tt & 1;
      int nt = wave * 8 + (tt >> 1);
      v8f acc = {};
      for (int k0 = 0; k0 < K; k0 += 32) {
        v16h a = load_a16(Abuf + mt * 16 * K + k0, K, lane);
        v16h b = load_bp(W + ((size_t)((k0 >> 5) * 64 + nt) << 9), lane);
        acc = __builtin_amdgcn_wmma_f32_16x16x32_f16(false, a, false, b, (short)0, acc,
                                                     false, false);
      }
      int col = nt * 16 + (lane & 15);
#pragma unroll
      for (int r = 0; r < 8; ++r) {
        int row = mt * 16 + r + ((lane >> 4) << 3);
        gates[row * 1024 + col] = acc[r];
      }
    }
    __syncthreads();

    // activations (i,f,g,o ordering) + build next A in LDS
    float* cptr = cst + cell * N_ * H_;
    _Float16* hptr = hst + cell * N_ * H_;
    const _Float16* hnext = hst + (cell + 1) * N_ * H_;  // valid for cell<2
    for (int idx = tid; idx < N_ * H_; idx += 256) {
      int n = idx >> 8, hh = idx & 255;
      const float* g = gates + n * 1024;
      float gi = g[hh]        + bc[hh];
      float gf = g[256 + hh]  + bc[256 + hh];
      float gg = g[512 + hh]  + bc[512 + hh];
      float go = g[768 + hh]  + bc[768 + hh];
      float c  = cptr[idx];
      float cn = sigm(gf) * c + sigm(gi) * tanhf(gg);
      float hn = sigm(go) * tanhf(cn);
      cptr[idx] = cn;
      _Float16 hn16 = (_Float16)hn;
      hptr[idx] = hn16;
      if (cell < 2) {
        Abuf[n * 512 + hh]       = hn16;        // next A = [h_new | h_next_old]
        Abuf[n * 512 + 256 + hh] = hnext[idx];
      } else {
        Abuf[n * 256 + hh] = hn16;              // A for q projection
      }
    }
    __syncthreads();
  }

  // q(32x256) = h2(32x256) @ WqT(256x256) + bq; 8 waves x (2 Mtiles x 2 Ntiles); NT=16
  for (int tt = 0; tt < 4; ++tt) {
    int mt = tt & 1;
    int nt = wave * 2 + (tt >> 1);
    v8f acc = {};
    for (int k0 = 0; k0 < 256; k0 += 32) {
      v16h a = load_a16(Abuf + mt * 16 * 256 + k0, 256, lane);
      v16h b = load_bp(WqP + ((size_t)((k0 >> 5) * 16 + nt) << 9), lane);
      acc = __builtin_amdgcn_wmma_f32_16x16x32_f16(false, a, false, b, (short)0, acc,
                                                   false, false);
    }
    int col = nt * 16 + (lane & 15);
    float bb = bq[col];
#pragma unroll
    for (int r = 0; r < 8; ++r) {
      int n = mt * 16 + r + ((lane >> 4) << 3);
      qout[n * 256 + col] = (_Float16)(acc[r] + bb);
    }
  }
}

// ------------------- attention + output head (one block per batch row) -------------------
__global__ void attention_out_kernel(const _Float16* __restrict__ key,
                                     const _Float16* __restrict__ val,
                                     const _Float16* __restrict__ q,
                                     const int* __restrict__ seq_lens,
                                     _Float16* __restrict__ ctx,
                                     const _Float16* __restrict__ Wo1T,
                                     const float* __restrict__ bo1,
                                     const float* __restrict__ embedding,
                                     const float* __restrict__ bo2,
                                     float* __restrict__ out /* null: skip head */) {
  __shared__ float sc[L_];
  __shared__ float red[256];
  __shared__ __align__(16) _Float16 qs[Q_];
  __shared__ float ctxs[H_];
  __shared__ float hid[H_];
  const int n = blockIdx.x, tid = threadIdx.x;
  qs[tid] = q[n * Q_ + tid];
  __syncthreads();
  // scores (vectorized b128 loads from L2-resident f16 keys)
  const h8v* qs8 = (const h8v*)qs;
  for (int l = tid; l < L_; l += 256) {
    const h8v* kp8 = (const h8v*)(key + ((size_t)n * L_ + l) * Q_);
    float acc = 0.f;
#pragma unroll 4
    for (int kk = 0; kk < 32; ++kk) {
      h8v kv = kp8[kk];
      h8v qv = qs8[kk];
#pragma unroll
      for (int j = 0; j < 8; ++j) acc += (float)kv[j] * (float)qv[j];
    }
    sc[l] = acc;
  }
  __syncthreads();
  // softmax over full L (then mask, clamp 1e-9, renormalize — faithful to ref)
  float m = -3.0e38f;
  for (int l = tid; l < L_; l += 256) m = fmaxf(m, sc[l]);
  red[tid] = m; __syncthreads();
  for (int s = 128; s > 0; s >>= 1) { if (tid < s) red[tid] = fmaxf(red[tid], red[tid + s]); __syncthreads(); }
  m = red[0]; __syncthreads();
  float s1 = 0.f;
  for (int l = tid; l < L_; l += 256) { float p = __expf(sc[l] - m); sc[l] = p; s1 += p; }
  red[tid] = s1; __syncthreads();
  for (int s = 128; s > 0; s >>= 1) { if (tid < s) red[tid] += red[tid + s]; __syncthreads(); }
  float S = red[0]; __syncthreads();
  const int len = seq_lens[n];
  float invS = 1.f / S;
  float s2 = 0.f;
  for (int l = tid; l < L_; l += 256) {
    float a = (l < len) ? sc[l] * invS : 0.f;
    a = fmaxf(a, 1e-9f);
    sc[l] = a; s2 += a;
  }
  red[tid] = s2; __syncthreads();
  for (int s = 128; s > 0; s >>= 1) { if (tid < s) red[tid] += red[tid + s]; __syncthreads(); }
  float inv2 = 1.f / red[0]; __syncthreads();
  // ctx[h] = sum_l attn[l] * val[n][l][h]  (coalesced across h=tid)
  const int h = tid;
  const _Float16* vp = val + (size_t)n * L_ * H_ + h;
  float acc = 0.f;
#pragma unroll 4
  for (int l = 0; l < L_; ++l) acc += sc[l] * (float)vp[(size_t)l * H_];
  float cv = acc * inv2;
  ctxs[h] = cv;
  ctx[n * H_ + h] = (_Float16)cv;
  __syncthreads();
  if (out != nullptr) {
    // hid[j] = bo1[j] + [ctx|q] . Wo1T[:,j]
    const int j = tid;
    float hacc = bo1[j];
#pragma unroll 4
    for (int k = 0; k < 256; ++k) {
      hacc += ctxs[k] * (float)Wo1T[k * 256 + j];
      hacc += (float)qs[k] * (float)Wo1T[(256 + k) * 256 + j];
    }
    hid[j] = hacc;
    __syncthreads();
    if (tid < V_) {
      const int v = tid;
      float lacc = bo2[v];
      const float* ep = embedding + v * H_;
#pragma unroll 4
      for (int k = 0; k < 256; ++k) lacc += hid[k] * ep[k];
      out[n * V_ + v] = lacc;
    }
  }
}

// ------------------- host driver -------------------
extern "C" void kernel_launch(void* const* d_in, const int* in_sizes, int n_in,
                              void* d_out, int out_size, void* d_ws, size_t ws_size,
                              hipStream_t stream) {
  (void)in_sizes; (void)n_in; (void)out_size; (void)ws_size;
  const float* seqs      = (const float*)d_in[0];
  const int*   seq_lens  = (const int*)d_in[1];
  const int*   label_in  = (const int*)d_in[2];
  const float* embedding = (const float*)d_in[3];
  const float* inith     = (const float*)d_in[4];
  const float* initc     = (const float*)d_in[5];
  const float* W_ih0 = (const float*)d_in[6];
  const float* W_hh0 = (const float*)d_in[7];
  const float* b_ih0 = (const float*)d_in[8];
  const float* b_hh0 = (const float*)d_in[9];
  const float* W_ih1 = (const float*)d_in[10];
  const float* W_hh1 = (const float*)d_in[11];
  const float* b_ih1 = (const float*)d_in[12];
  const float* b_hh1 = (const float*)d_in[13];
  const float* W_ih2 = (const float*)d_in[14];
  const float* W_hh2 = (const float*)d_in[15];
  const float* b_ih2 = (const float*)d_in[16];
  const float* b_hh2 = (const float*)d_in[17];
  const float* Wq  = (const float*)d_in[18];
  const float* bq  = (const float*)d_in[19];
  const float* Wk  = (const float*)d_in[20];
  const float* bk  = (const float*)d_in[21];
  const float* Wv  = (const float*)d_in[22];
  const float* bv  = (const float*)d_in[23];
  const float* Wo1 = (const float*)d_in[24];
  const float* bo1 = (const float*)d_in[25];
  const float* bo2 = (const float*)d_in[26];
  float* out = (float*)d_out;

  // workspace carve-up (256B aligned)
  char* ws = (char*)d_ws;
  size_t off = 0;
  auto take = [&](size_t bytes) -> char* {
    char* p = ws + off;
    off = (off + bytes + 255) & ~(size_t)255;
    return p;
  };
  _Float16* wkP    = (_Float16*)take((size_t)IN_ * Q_ * 2);       // packed B frags
  _Float16* wvP    = (_Float16*)take((size_t)IN_ * H_ * 2);
  _Float16* w0     = (_Float16*)take((size_t)768 * 1024 * 2);
  _Float16* w1     = (_Float16*)take((size_t)512 * 1024 * 2);
  _Float16* w2     = (_Float16*)take((size_t)512 * 1024 * 2);
  _Float16* wqP    = (_Float16*)take((size_t)256 * 256 * 2);
  _Float16* wo1T   = (_Float16*)take((size_t)512 * 256 * 2);
  float*    bc0    = (float*)take(1024 * 4);
  float*    bc1    = (float*)take(1024 * 4);
  float*    bc2    = (float*)take(1024 * 4);
  _Float16* embseq = (_Float16*)take((size_t)T_ * N_ * H_ * 2);
  _Float16* key    = (_Float16*)take((size_t)N_ * L_ * Q_ * 2);
  _Float16* val    = (_Float16*)take((size_t)N_ * L_ * H_ * 2);
  _Float16* hst    = (_Float16*)take((size_t)3 * N_ * H_ * 2);
  float*    cst    = (float*)take((size_t)3 * N_ * H_ * 4);
  _Float16* ctxb   = (_Float16*)take((size_t)N_ * H_ * 2);
  _Float16* qb     = (_Float16*)take((size_t)N_ * Q_ * 2);

  auto g = [](int n) { return (n + 255) / 256; };

  // fragment-native packed weights (f16)
  pack_b_kernel<<<g(512 * 256), 256, 0, stream>>>(wkP, Wk, 512, Wk, 0, 256);
  pack_b_kernel<<<g(512 * 256), 256, 0, stream>>>(wvP, Wv, 512, Wv, 0, 256);
  pack_b_kernel<<<g(768 * 1024), 256, 0, stream>>>(w0, W_ih0, 512, W_hh0, 256, 1024);
  pack_b_kernel<<<g(512 * 1024), 256, 0, stream>>>(w1, W_ih1, 256, W_hh1, 256, 1024);
  pack_b_kernel<<<g(512 * 1024), 256, 0, stream>>>(w2, W_ih2, 256, W_hh2, 256, 1024);
  pack_b_kernel<<<g(256 * 256), 256, 0, stream>>>(wqP, Wq, 256, Wq, 0, 256);
  transpose_f16_kernel<<<g(256 * 512), 256, 0, stream>>>(wo1T, Wo1, 256, 512);
  add_bias_kernel<<<4, 256, 0, stream>>>(bc0, b_ih0, b_hh0, 1024);
  add_bias_kernel<<<4, 256, 0, stream>>>(bc1, b_ih1, b_hh1, 1024);
  add_bias_kernel<<<4, 256, 0, stream>>>(bc2, b_ih2, b_hh2, 1024);
  emb_gather_kernel<<<g(T_ * N_ * H_), 256, 0, stream>>>(embseq, embedding, label_in);
  state_init_kernel<<<g(3 * N_ * H_), 256, 0, stream>>>(hst, cst, inith, initc);
  q_init_kernel<<<N_, 256, 0, stream>>>(qb, inith + 2 * H_, Wq, bq);

  // KV projection: 4096 M-tiles, 8 waves/block
  kv_proj_kernel<<<512, 256, 0, stream>>>(seqs, wkP, wvP, bk, bv, key, val);

  // initial context from q0 (no output head)
  attention_out_kernel<<<N_, 256, 0, stream>>>(key, val, qb, seq_lens, ctxb,
                                               wo1T, bo1, embedding, bo2, nullptr);

  const int lstm_smem = 32 * 768 * 2 + 32 * 1024 * 4;  // 176 KB (within 320 KB WGP LDS)
  hipFuncSetAttribute(reinterpret_cast<const void*>(lstm_step_kernel),
                      hipFuncAttributeMaxDynamicSharedMemorySize, lstm_smem);

  for (int t = 0; t < T_; ++t) {
    lstm_step_kernel<<<1, 256, lstm_smem, stream>>>(ctxb, embseq + (size_t)t * N_ * H_, hst, cst,
                                                    w0, w1, w2, wqP, bc0, bc1, bc2, bq, qb);
    attention_out_kernel<<<N_, 256, 0, stream>>>(key, val, qb, seq_lens, ctxb,
                                                 wo1T, bo1, embedding, bo2,
                                                 out + (size_t)t * N_ * V_);
  }
}
